// ModelBlock_16578573763177
// MI455X (gfx1250) — matrix operand order repaired and checked by reading, not compiled, not run on hardware
//
#include <hip/hip_runtime.h>
#include <hip/hip_bf16.h>

// ---------------------------------------------------------------------------
// Mamba2 block for gfx1250 (MI455X). GEMMs via v_wmma_f32_16x16x32_bf16 with
// double-buffered LDS and global_load_async_to_lds_b128 staging (ASYNCcnt).
// Shapes: B=4, L=2048, H_MODEL=1024, D_INNER=2048, CONV_DIM=2176,
// D_IN_PROJ=4232 (padded to 4288), NHEADS=8, HEADDIM=256, N_STATE=64.
// ---------------------------------------------------------------------------

#define BATCH    4
#define SEQLEN   2048
#define HMODEL   1024
#define DINNER   2048
#define CONVDIM  2176
#define NPROJ    4232
#define NPAD     4288      // NPROJ padded up to a multiple of 64
#define NHEADS   8
#define HEADDIM  256
#define NSTATE   64
#define DCONV    16
#define MROWS    (BATCH * SEQLEN)   // 8192

typedef __attribute__((ext_vector_type(8)))  float          v8f;
typedef __attribute__((ext_vector_type(16))) __bf16         v16bf;
typedef __attribute__((ext_vector_type(8)))  unsigned short ushort8;
typedef __attribute__((ext_vector_type(16))) unsigned short ushort16;

static __device__ __forceinline__ unsigned short f2bf(float f) {
    unsigned int u = __float_as_uint(f);
    unsigned int r = u + 0x7FFFu + ((u >> 16) & 1u);   // round-to-nearest-even
    return (unsigned short)(r >> 16);
}

static __device__ __forceinline__ v16bf as_v16bf(ushort16 u) {
    union { ushort16 u; v16bf b; } x; x.u = u; return x.b;
}

static __device__ __forceinline__ float sigmoidf_(float x) {
    return 1.0f / (1.0f + __expf(-x));
}

// LDS workgroup-relative byte offset of a generic pointer to __shared__ data.
// ISA: generic LDS address low 32 bits == LDS offset (addr[31:0] truncation).
static __device__ __forceinline__ unsigned lds_off(const void* p) {
    return (unsigned)(unsigned long long)p;
}

// async global -> LDS copy of 16 bytes (tracked by ASYNCcnt)
static __device__ __forceinline__ void async_g2l_b128(unsigned ldsByteOff,
                                                      const void* gaddr) {
    asm volatile("global_load_async_to_lds_b128 %0, %1, off"
                 :: "v"(ldsByteOff), "v"(gaddr) : "memory");
}

static __device__ __forceinline__ void wait_asynccnt0() {
    asm volatile("s_wait_asynccnt 0x0" ::: "memory");
}

// ---------------------------------------------------------------------------
// fp32 -> bf16 conversion
// ---------------------------------------------------------------------------
__global__ void cvt_bf16_kernel(const float* __restrict__ src,
                                unsigned short* __restrict__ dst, int n) {
    int i = blockIdx.x * blockDim.x + threadIdx.x;
    if (i < n) dst[i] = f2bf(src[i]);
}

// W_in [1024][4232] fp32 -> bf16 [1024][4288], zero-padded columns
__global__ void cvt_pad_kernel(const float* __restrict__ src,
                               unsigned short* __restrict__ dst,
                               int rows, int cols, int colsPad) {
    int i = blockIdx.x * blockDim.x + threadIdx.x;
    int total = rows * colsPad;
    if (i >= total) return;
    int r = i / colsPad, c = i - r * colsPad;
    float v = (c < cols) ? src[(size_t)r * cols + c] : 0.0f;
    dst[i] = f2bf(v);
}

// ---------------------------------------------------------------------------
// WMMA bf16 GEMM:  C[M][N] = A[M][K] * B[K][N]
// M % 256 == 0, N % 64 == 0, K % 32 == 0.  256 threads = 8 waves/block.
// Block tile 256x64; each wave owns 32 rows x 64 cols = 2x4 WMMA tiles
// (8 x v_wmma_f32_16x16x32_bf16 per K-step). Double-buffered LDS: the
// async global->LDS copy of tile k+1 overlaps the WMMA block of tile k.
// LDS row stride 40 ushorts (80B) keeps fragment gathers conflict-free.
// ---------------------------------------------------------------------------
#define GBM 256
#define GBN 64
#define GBK 32
#define ASTR 40
#define BSTR 40

__global__ __launch_bounds__(256)
void gemm_bf16_kernel(const unsigned short* __restrict__ A,
                      const unsigned short* __restrict__ B,
                      float* __restrict__ C, int M, int N, int K) {
    __shared__ __align__(16) unsigned short As[2][GBM * ASTR];
    __shared__ __align__(16) unsigned short Bs[2][GBN * BSTR];

    const int tid  = threadIdx.x;
    const int lane = tid & 31;
    const int wave = tid >> 5;       // 0..7 -> wave's M offset /32
    const int bm   = blockIdx.y * GBM;
    const int bn   = blockIdx.x * GBN;

    v8f acc[2][4] = {};

    // staging assignments: thread t stages A row t (64B = 4 async b128)
    // and one 16B segment of B (row bk, cols bs*8..bs*8+7), transposed.
    const int bk = tid >> 3;         // 0..31 : B row (k) within tile
    const int bs = tid & 7;          // 0..7  : 8-col segment

    const unsigned short* aRow = A + (size_t)(bm + tid) * K;
    const unsigned short* bptr = B + (size_t)bk * N + bn + bs * 8;

    const unsigned aLds[2] = { lds_off(&As[0][tid * ASTR]),
                               lds_off(&As[1][tid * ASTR]) };

    const int fr = lane & 15;        // fragment row/col within 16
    const int kh = lane >> 4;        // fragment K-half select

    const int nk = K >> 5;

    // ---- prologue: stage tile 0 into buffer 0 ----
    {
        const unsigned short* ag = aRow;
        unsigned l = aLds[0];
        async_g2l_b128(l,       ag);
        async_g2l_b128(l + 16u, ag + 8);
        async_g2l_b128(l + 32u, ag + 16);
        async_g2l_b128(l + 48u, ag + 24);
        ushort8 bv0 = *(const ushort8*)bptr;
        #pragma unroll
        for (int i = 0; i < 8; ++i)
            Bs[0][(bs * 8 + i) * BSTR + bk] = bv0[i];
    }
    wait_asynccnt0();
    __syncthreads();

    for (int kt = 0; kt < nk; ++kt) {
        const int cur = kt & 1;
        const int nxt = cur ^ 1;
        const bool hasNext = (kt + 1) < nk;

        // ---- issue async A copy + B load for tile k+1 (other parity) ----
        ushort8 bv;
        if (hasNext) {
            const unsigned short* ag = aRow + (size_t)(kt + 1) * 32;
            unsigned l = aLds[nxt];
            async_g2l_b128(l,       ag);
            async_g2l_b128(l + 16u, ag + 8);
            async_g2l_b128(l + 32u, ag + 16);
            async_g2l_b128(l + 48u, ag + 24);
            bv = *(const ushort8*)(bptr + (size_t)(kt + 1) * 32 * N);
            if (kt + 2 < nk) {
                __builtin_prefetch(aRow + (size_t)(kt + 2) * 32, 0, 1);
                __builtin_prefetch(bptr + (size_t)(kt + 2) * 32 * N, 0, 1);
            }
        }

        // ---- compute tile k from current parity ----
        // A 16x32 bf16 frag: lanes 0-15 row M, K=0..7 & 16..23 ;
        //                    lanes 16-31            K=8..15 & 24..31
        ushort16 af[2];
        #pragma unroll
        for (int mt = 0; mt < 2; ++mt) {
            int row = wave * 32 + mt * 16 + fr;
            ushort8 lo = *(const ushort8*)&As[cur][row * ASTR + kh * 8];
            ushort8 hi = *(const ushort8*)&As[cur][row * ASTR + 16 + kh * 8];
            af[mt] = __builtin_shufflevector(lo, hi,
                0,1,2,3,4,5,6,7,8,9,10,11,12,13,14,15);
        }
        // B 32x16 bf16 frag: lanes 0-15 col N, K=0..7 & 16..23 ; lanes 16-31 K=8..15 & 24..31
        ushort16 bf4[4];
        #pragma unroll
        for (int nt = 0; nt < 4; ++nt) {
            int col = nt * 16 + fr;
            ushort8 lo = *(const ushort8*)&Bs[cur][col * BSTR + kh * 8];
            ushort8 hi = *(const ushort8*)&Bs[cur][col * BSTR + 16 + kh * 8];
            bf4[nt] = __builtin_shufflevector(lo, hi,
                0,1,2,3,4,5,6,7,8,9,10,11,12,13,14,15);
        }

        #pragma unroll
        for (int mt = 0; mt < 2; ++mt)
            #pragma unroll
            for (int nt = 0; nt < 4; ++nt)
                acc[mt][nt] = __builtin_amdgcn_wmma_f32_16x16x32_bf16(
                    false, as_v16bf(af[mt]),
                    false, as_v16bf(bf4[nt]),
                    (short)0, acc[mt][nt], false, false);

        // ---- finish staging tile k+1, then one barrier ----
        if (hasNext) {
            #pragma unroll
            for (int i = 0; i < 8; ++i)
                Bs[nxt][(bs * 8 + i) * BSTR + bk] = bv[i];
            wait_asynccnt0();
        }
        __syncthreads();
    }

    // ---- store C: lane L -> col = L&15 ; VGPR v -> row = v + 8*(L>>4) ----
    #pragma unroll
    for (int mt = 0; mt < 2; ++mt) {
        #pragma unroll
        for (int nt = 0; nt < 4; ++nt) {
            float* cp = C + (size_t)(bm + wave * 32 + mt * 16 + 8 * kh) * N
                          + bn + nt * 16 + fr;
            #pragma unroll
            for (int v = 0; v < 8; ++v)
                cp[(size_t)v * N] = acc[mt][nt][v];
        }
    }
}

// ---------------------------------------------------------------------------
// dt = softplus(zxbcdt[..., -8:] + dt_bias) ; dA = exp(dt * (-exp(A_log)))
// ---------------------------------------------------------------------------
__global__ void dt_kernel(const float* __restrict__ zx,
                          const float* __restrict__ dt_bias,
                          const float* __restrict__ A_log,
                          float* __restrict__ dt, float* __restrict__ dA) {
    int i = blockIdx.x * blockDim.x + threadIdx.x;   // MROWS*NHEADS
    if (i >= MROWS * NHEADS) return;
    int bl = i >> 3, h = i & 7;
    float raw = zx[(size_t)bl * NPAD + (NPROJ - NHEADS) + h] + dt_bias[h];
    float sp  = (raw > 20.0f) ? raw : log1pf(__expf(raw));
    float A   = -__expf(A_log[h]);
    dt[i] = sp;
    dA[i] = __expf(sp * A);
}

// ---------------------------------------------------------------------------
// depthwise causal conv (16 taps) + bias + SiLU over xBC channels
// one thread per (batch, channel), sliding window in registers
// ---------------------------------------------------------------------------
__global__ __launch_bounds__(256)
void conv_silu_kernel(const float* __restrict__ zx,
                      const float* __restrict__ conv_w,
                      const float* __restrict__ conv_b,
                      float* __restrict__ xbc) {
    int c = blockIdx.x * blockDim.x + threadIdx.x;
    int b = blockIdx.y;
    if (c >= CONVDIM) return;

    float w[DCONV];
    #pragma unroll
    for (int j = 0; j < DCONV; ++j) w[j] = conv_w[c * DCONV + j];
    float bias = conv_b[c];

    float hist[DCONV];
    #pragma unroll
    for (int j = 0; j < DCONV; ++j) hist[j] = 0.0f;

    const float* in  = zx  + (size_t)b * SEQLEN * NPAD + DINNER + c;
    float*       out = xbc + (size_t)b * SEQLEN * CONVDIM + c;

    for (int l = 0; l < SEQLEN; ++l) {
        #pragma unroll
        for (int j = 0; j < DCONV - 1; ++j) hist[j] = hist[j + 1];
        hist[DCONV - 1] = in[(size_t)l * NPAD];
        float acc = bias;
        #pragma unroll
        for (int j = 0; j < DCONV; ++j) acc += w[j] * hist[j];
        out[(size_t)l * CONVDIM] = acc * sigmoidf_(acc);
    }
}

// ---------------------------------------------------------------------------
// recurrent selective scan: one block per (batch, head); thread p owns
// state row h[p][0..63] in registers. y = h@C + D*x.
// ---------------------------------------------------------------------------
__global__ __launch_bounds__(256)
void scan_kernel(const float* __restrict__ xbc,
                 const float* __restrict__ dt,
                 const float* __restrict__ dA,
                 const float* __restrict__ Dvec,
                 const float* __restrict__ init_states,
                 float* __restrict__ ybuf) {
    const int h = blockIdx.x;        // 0..7
    const int b = blockIdx.y;        // 0..3
    const int p = threadIdx.x;       // 0..255

    __shared__ float Bl[NSTATE];
    __shared__ float Cl[NSTATE];

    float st[NSTATE];
    #pragma unroll
    for (int n = 0; n < NSTATE; ++n)
        st[n] = init_states[((size_t)h * HEADDIM + p) * NSTATE + n];

    const float Dh = Dvec[h];

    for (int t = 0; t < SEQLEN; ++t) {
        const size_t row = (size_t)b * SEQLEN + t;
        if (p < NSTATE)
            Bl[p] = xbc[row * CONVDIM + DINNER + p];
        else if (p < 2 * NSTATE)
            Cl[p - NSTATE] = xbc[row * CONVDIM + DINNER + NSTATE + (p - NSTATE)];
        __syncthreads();

        float xv  = xbc[row * CONVDIM + h * HEADDIM + p];
        float dtv = dt[row * NHEADS + h];
        float dAv = dA[row * NHEADS + h];
        float coef = dtv * xv;

        float y = 0.0f;
        #pragma unroll
        for (int n = 0; n < NSTATE; ++n) {
            float s = st[n] * dAv + coef * Bl[n];
            st[n] = s;
            y += s * Cl[n];
        }
        y += Dh * xv;
        ybuf[row * DINNER + h * HEADDIM + p] = y;
        __syncthreads();
    }
}

// ---------------------------------------------------------------------------
// y = y * silu(z); RMSNorm over D_INNER; * norm_w; -> bf16
// one block per token row; 256 threads x 8 elements each
// ---------------------------------------------------------------------------
__global__ __launch_bounds__(256)
void gate_norm_kernel(const float* __restrict__ ybuf,
                      const float* __restrict__ zx,
                      const float* __restrict__ norm_w,
                      unsigned short* __restrict__ ynorm_bf) {
    const int row = blockIdx.x;
    const int tid = threadIdx.x;
    __shared__ float red[256];

    float vk[8];
    float ss = 0.0f;
    #pragma unroll
    for (int k = 0; k < 8; ++k) {
        int c = tid + k * 256;
        float z = zx[(size_t)row * NPAD + c];
        float y = ybuf[(size_t)row * DINNER + c];
        float v = y * (z * sigmoidf_(z));
        vk[k] = v;
        ss += v * v;
    }
    red[tid] = ss;
    __syncthreads();
    for (int s = 128; s > 0; s >>= 1) {
        if (tid < s) red[tid] += red[tid + s];
        __syncthreads();
    }
    float scale = rsqrtf(red[0] / (float)DINNER + 1e-5f);
    #pragma unroll
    for (int k = 0; k < 8; ++k) {
        int c = tid + k * 256;
        ynorm_bf[(size_t)row * DINNER + c] = f2bf(vk[k] * scale * norm_w[c]);
    }
}

// t = y / sqrt(1 + y*y)  -> bf16
__global__ void tanhish_kernel(const float* __restrict__ y2,
                               unsigned short* __restrict__ tbf, int n) {
    int i = blockIdx.x * blockDim.x + threadIdx.x;
    if (i >= n) return;
    float v = y2[i];
    tbf[i] = f2bf(v * rsqrtf(1.0f + v * v));
}

// out = g3 + b_nl + x
__global__ void final_add_kernel(const float* __restrict__ g3,
                                 const float* __restrict__ b_nl,
                                 const float* __restrict__ x,
                                 float* __restrict__ out, int n) {
    int i = blockIdx.x * blockDim.x + threadIdx.x;
    if (i >= n) return;
    out[i] = g3[i] + b_nl[i & (HMODEL - 1)] + x[i];
}

// ---------------------------------------------------------------------------
// host-side orchestration
// ---------------------------------------------------------------------------
static inline size_t align256(size_t x) { return (x + 255) & ~(size_t)255; }

extern "C" void kernel_launch(void* const* d_in, const int* in_sizes, int n_in,
                              void* d_out, int out_size, void* d_ws, size_t ws_size,
                              hipStream_t stream) {
    const float* x        = (const float*)d_in[0];
    const float* W_in     = (const float*)d_in[1];
    const float* conv_w   = (const float*)d_in[2];
    const float* conv_b   = (const float*)d_in[3];
    const float* dt_bias  = (const float*)d_in[4];
    const float* A_log    = (const float*)d_in[5];
    const float* Dvec     = (const float*)d_in[6];
    const float* init_st  = (const float*)d_in[7];
    const float* norm_w   = (const float*)d_in[8];
    const float* W_out    = (const float*)d_in[9];
    const float* W_nl     = (const float*)d_in[10];
    const float* b_nl     = (const float*)d_in[11];
    float* out            = (float*)d_out;

    char* ws = (char*)d_ws;
    size_t off = 0;
    unsigned short* x_bf   = (unsigned short*)(ws + off); off += align256((size_t)MROWS * HMODEL * 2);
    unsigned short* Win_bf = (unsigned short*)(ws + off); off += align256((size_t)HMODEL * NPAD * 2);
    float*          zx     = (float*)(ws + off);          off += align256((size_t)MROWS * NPAD * 4);
    float*          xbc    = (float*)(ws + off);          off += align256((size_t)MROWS * CONVDIM * 4);
    float*          dt     = (float*)(ws + off);          off += align256((size_t)MROWS * NHEADS * 4);
    float*          dA     = (float*)(ws + off);          off += align256((size_t)MROWS * NHEADS * 4);
    float*          ybuf   = (float*)(ws + off);          off += align256((size_t)MROWS * DINNER * 4);
    unsigned short* yn_bf  = (unsigned short*)(ws + off); off += align256((size_t)MROWS * DINNER * 2);
    unsigned short* Wo_bf  = (unsigned short*)(ws + off); off += align256((size_t)DINNER * HMODEL * 2);
    float*          y2     = (float*)(ws + off);          off += align256((size_t)MROWS * HMODEL * 4);
    unsigned short* t_bf   = (unsigned short*)(ws + off); off += align256((size_t)MROWS * HMODEL * 2);
    unsigned short* Wnl_bf = (unsigned short*)(ws + off); off += align256((size_t)HMODEL * HMODEL * 2);
    float*          g3     = (float*)(ws + off);          off += align256((size_t)MROWS * HMODEL * 4);
    (void)ws_size; (void)in_sizes; (void)n_in; (void)out_size;

    const int T = 256;

    // precision conversions
    {
        int n = MROWS * HMODEL;
        cvt_bf16_kernel<<<(n + T - 1) / T, T, 0, stream>>>(x, x_bf, n);
    }
    {
        int n = HMODEL * NPAD;
        cvt_pad_kernel<<<(n + T - 1) / T, T, 0, stream>>>(W_in, Win_bf, HMODEL, NPROJ, NPAD);
    }
    {
        int n = DINNER * HMODEL;
        cvt_bf16_kernel<<<(n + T - 1) / T, T, 0, stream>>>(W_out, Wo_bf, n);
    }
    {
        int n = HMODEL * HMODEL;
        cvt_bf16_kernel<<<(n + T - 1) / T, T, 0, stream>>>(W_nl, Wnl_bf, n);
    }

    // zxbcdt = x @ W_in   [8192 x 4288] (K = 1024)
    {
        dim3 grid(NPAD / GBN, MROWS / GBM);
        gemm_bf16_kernel<<<grid, T, 0, stream>>>(x_bf, Win_bf, zx, MROWS, NPAD, HMODEL);
    }

    // dt / dA
    {
        int n = MROWS * NHEADS;
        dt_kernel<<<(n + T - 1) / T, T, 0, stream>>>(zx, dt_bias, A_log, dt, dA);
    }

    // depthwise conv + SiLU
    {
        dim3 grid((CONVDIM + T - 1) / T, BATCH);
        conv_silu_kernel<<<grid, T, 0, stream>>>(zx, conv_w, conv_b, xbc);
    }

    // recurrent scan
    {
        dim3 grid(NHEADS, BATCH);
        scan_kernel<<<grid, T, 0, stream>>>(xbc, dt, dA, Dvec, init_st, ybuf);
    }

    // gate + RMSNorm -> bf16
    gate_norm_kernel<<<MROWS, T, 0, stream>>>(ybuf, zx, norm_w, yn_bf);

    // y2 = ynorm @ W_out   [8192 x 1024] (K = 2048)
    {
        dim3 grid(HMODEL / GBN, MROWS / GBM);
        gemm_bf16_kernel<<<grid, T, 0, stream>>>(yn_bf, Wo_bf, y2, MROWS, HMODEL, DINNER);
    }

    // t = y2 / sqrt(1 + y2^2) -> bf16
    {
        int n = MROWS * HMODEL;
        tanhish_kernel<<<(n + T - 1) / T, T, 0, stream>>>(y2, t_bf, n);
    }

    // g3 = t @ W_nl   [8192 x 1024] (K = 1024)
    {
        dim3 grid(HMODEL / GBN, MROWS / GBM);
        gemm_bf16_kernel<<<grid, T, 0, stream>>>(t_bf, Wnl_bf, g3, MROWS, HMODEL, HMODEL);
    }

    // out = g3 + b_nl + x
    {
        int n = MROWS * HMODEL;
        final_add_kernel<<<(n + T - 1) / T, T, 0, stream>>>(g3, b_nl, x, out, n);
    }
}
